// WindowAttention_884763263594
// MI455X (gfx1250) — compile-verified, matrix operand order checked by reading, not run on hardware
//
#include <hip/hip_runtime.h>

#define NTOK 49
#define NPAD 64
#define HEADS 4
#define HD 32
#define CCH 128
#define NWMASK 1024
#define THREADS 256

typedef _Float16 h8   __attribute__((ext_vector_type(8)));
typedef _Float16 v16h __attribute__((ext_vector_type(16)));
typedef float    v8f  __attribute__((ext_vector_type(8)));
typedef float    f4   __attribute__((ext_vector_type(4)));
typedef unsigned int u32x4 __attribute__((ext_vector_type(4)));
typedef int      i32x4 __attribute__((ext_vector_type(4)));
typedef int      i32x8 __attribute__((ext_vector_type(8)));

// 16-bit WMMA A/B fragment from row-major f16 LDS.
// Lane L holds row/col (r0 + L%16); lanes 0-15 take K = {k0..k0+7} ∪ {k0+16..k0+23},
// lanes 16-31 take K = {k0+8..k0+15} ∪ {k0+24..k0+31}; two contiguous 16B chunks.
__device__ __forceinline__ v16h ld_frag_lds(const _Float16* p, int stride, int r0, int k0) {
    const int lane = threadIdx.x & 31;
    const _Float16* q = p + (r0 + (lane & 15)) * stride + k0 + ((lane >> 4) << 3);
    h8 lo = *(const h8*)q;
    h8 hi = *(const h8*)(q + 16);
    return __builtin_shufflevector(lo, hi, 0,1,2,3,4,5,6,7,8,9,10,11,12,13,14,15);
}

// Same fragment from a pre-converted f16 weight matrix in global memory.
__device__ __forceinline__ v16h ld_frag_g16(const _Float16* __restrict__ w, int ldc, int n0, int k0) {
    const int lane = threadIdx.x & 31;
    const _Float16* q = w + (size_t)(n0 + (lane & 15)) * ldc + k0 + ((lane >> 4) << 3);
    h8 lo = *(const h8*)q;
    h8 hi = *(const h8*)(q + 16);
    return __builtin_shufflevector(lo, hi, 0,1,2,3,4,5,6,7,8,9,10,11,12,13,14,15);
}

__device__ __forceinline__ v8f wmma16(v16h a, v16h b, v8f c) {
    return __builtin_amdgcn_wmma_f32_16x16x32_f16(false, a, false, b, (short)0, c, false, false);
}

__device__ __forceinline__ h8 pack8(v8f acc, float s) {
    h8 pk;
#pragma unroll
    for (int v = 0; v < 8; ++v) pk[v] = (_Float16)(acc[v] * s);
    return pk;
}

// ---------------- workspace pre-pass -----------------------------------------
#define WS_QW   0
#define WS_PW   98304
#define WS_BM   131072

__global__ __launch_bounds__(THREADS)
void prep_kernel(const float* __restrict__ qkv_w, const float* __restrict__ proj_w,
                 const float* __restrict__ tab,   const int* __restrict__ idx,
                 char* __restrict__ ws)
{
    _Float16* qwh = (_Float16*)(ws + WS_QW);
    _Float16* pwh = (_Float16*)(ws + WS_PW);
    float*    bm  = (float*)(ws + WS_BM);
    int i = blockIdx.x * THREADS + threadIdx.x;
    if (i < 384 * CCH) qwh[i] = (_Float16)qkv_w[i];
    if (i < CCH * CCH) pwh[i] = (_Float16)proj_w[i];
    if (i < HEADS * NTOK * NTOK) {
        int h = i / (NTOK * NTOK), e = i % (NTOK * NTOK);
        bm[i] = tab[idx[e] * HEADS + h];
    }
}

// ---------------- main fused window-attention kernel -------------------------
// LDS regions (aliased by lifetime; every alias pair is barrier-separated):
//   [0)      A: xh 64x136 f16 (P0-P1) ∪ ath 4x64x72 f16 (P2b-P3)   : 36864 B
//   [36864)  qkh 64x264 f16 (P1-P2)  [token][q 0..127 | k 128..255] : 33792 B
//   [70656)  vT 128x72 f16 (P1-P3)   [h*32+d][token]                : 18432 B
//   [89088)  B: xf32 49x128 f32 (TDM, P0) ∪ sc 4x64x68 f32 (P2-P2b)
//               ∪ oh 64x136 f16 (P3-P4)                             : 69632 B
// total 158720 B -> two workgroups per 320KB WGP.
#define SMEM_BYTES 158720
#define XF32_OFF   89088

__global__ __launch_bounds__(THREADS)
void win_attn_kernel(const float* __restrict__ x,
                     const float* __restrict__ proj_b,
                     const float* __restrict__ mask,
                     const char*  __restrict__ ws,
                     float*       __restrict__ out)
{
    extern __shared__ __align__(16) char smem_raw[];
    _Float16* xh   = (_Float16*)smem_raw;             // P0-P1
    _Float16* ath  = (_Float16*)smem_raw;             // P2b-P3 (aliases xh)
    _Float16* qkh  = (_Float16*)(smem_raw + 36864);   // P1-P2
    _Float16* vT   = (_Float16*)(smem_raw + 70656);   // P1-P3
    float*    xf32 = (float*)   (smem_raw + XF32_OFF);// P0 (TDM landing)
    float*    sc   = (float*)   (smem_raw + XF32_OFF);// P2-P2b
    _Float16* oh   = (_Float16*)(smem_raw + XF32_OFF);// P3-P4

    const _Float16* qwh = (const _Float16*)(ws + WS_QW);
    const _Float16* pwh = (const _Float16*)(ws + WS_PW);
    const float*    bm  = (const float*)   (ws + WS_BM);

    const int b    = blockIdx.x;
    const int tid  = threadIdx.x;
    const int lane = tid & 31;
    const int wv   = tid >> 5;
    const int lcol = lane & 15;              // D-tile column owned by this lane
    const int lrow = (lane >> 4) << 3;       // first of 8 contiguous D rows
    const float* xb    = x + (size_t)b * NTOK * CCH;
    const float* maskb = mask + (size_t)(b & (NWMASK - 1)) * NTOK * NTOK;

    if (tid == 0) { __builtin_prefetch(qwh, 0, 3); __builtin_prefetch(pwh, 0, 3); }

    // ---- P0a: Tensor Data Mover copies x[b] (49x128 fp32) into LDS.
    // D# group0: count=1, lds_addr, global_addr[56:0], type=2.
    // D# group1: data_size=4B, tensor 128x49 (dim0 stride 128), tile 128x49.
    if (wv == 0) {
        unsigned long long ga = (unsigned long long)(const void*)xb;
        u32x4 g0 = { 1u, (unsigned)XF32_OFF, (unsigned)(ga & 0xffffffffu),
                     (unsigned)(((ga >> 32) & 0x1ffffffu) | 0x80000000u) };
        i32x8 g1 = { 0x00020000,          // workgroup_mask=0, data_size=2 (4B)
                     (int)(CCH << 16),    // tensor_dim0 = 128 (low 16 in [31:16])
                     (int)(NTOK << 16),   // tensor_dim0 hi=0 | tensor_dim1 = 49
                     (int)(CCH << 16),    // tensor_dim1 hi=0 | tile_dim0 = 128
                     NTOK,                // tile_dim1 = 49, tile_dim2 = 0
                     CCH,                 // tensor_dim0_stride = 128
                     0, 0 };              // tensor_dim1_stride = 0 (2D)
        i32x4 gz4 = { 0, 0, 0, 0 };
        i32x8 gz8 = { 0, 0, 0, 0, 0, 0, 0, 0 };
        __builtin_amdgcn_tensor_load_to_lds(g0, g1, gz4, gz4, gz8, 0);
        __builtin_amdgcn_s_wait_tensorcnt(0);
    }
    __syncthreads();

    // ---- P0b: convert staged fp32 -> f16 xh (zero-pad rows/cols), 16B chunks.
    for (int i = tid; i < 64 * 17; i += THREADS) {      // 17 h8-chunks per row
        int r = i / 17, c8 = i % 17;
        h8 pk = {};
        if (r < NTOK && c8 < 16) {
            f4 lo = *(const f4*)(xf32 + r * CCH + c8 * 8);
            f4 hi = *(const f4*)(xf32 + r * CCH + c8 * 8 + 4);
#pragma unroll
            for (int v = 0; v < 4; ++v) { pk[v] = (_Float16)lo[v]; pk[4 + v] = (_Float16)hi[v]; }
        }
        *(h8*)(xh + r * 136 + c8 * 8) = pk;
    }
    __syncthreads();

    // ---- P1a: q/k tiles as W·Xᵀ (64 tiles, 8/wave) -> lane owns a token,
    // 8 contiguous channels -> single ds_store_b128 into qkh[token][ch].
    const float qscale = 0.17677669529663689f;  // 1/sqrt(32)
#pragma unroll
    for (int i = 0; i < 8; ++i) {
        int t = wv * 8 + i;
        int ot = t >> 2, tt = t & 3;
        v8f acc = {};
#pragma unroll
        for (int kt = 0; kt < 4; ++kt) {
            v16h a  = ld_frag_g16(qwh, CCH, ot * 16, kt * 32);
            v16h bb = ld_frag_lds(xh, 136, tt * 16, kt * 32);
            acc = wmma16(a, bb, acc);
        }
        int token = tt * 16 + lcol;
        int o0    = ot * 16 + lrow;          // 0..255 (q then k)
        *(h8*)(qkh + token * 264 + o0) = pack8(acc, (ot < 8) ? qscale : 1.f);
    }
    // ---- P1b: v tiles as X·Wᵀ (32 tiles, 4/wave) -> lane owns a channel,
    // 8 contiguous tokens -> single ds_store_b128 into vT[ch][token].
#pragma unroll
    for (int i = 0; i < 4; ++i) {
        int t = wv * 4 + i;
        int tt = t >> 3, vt = t & 7;
        v8f acc = {};
#pragma unroll
        for (int kt = 0; kt < 4; ++kt) {
            v16h a  = ld_frag_lds(xh, 136, tt * 16, kt * 32);
            v16h bb = ld_frag_g16(qwh, CCH, 256 + vt * 16, kt * 32);
            acc = wmma16(a, bb, acc);
        }
        *(h8*)(vT + (vt * 16 + lcol) * 72 + tt * 16 + lrow) = pack8(acc, 1.f);
    }
    __syncthreads();

    // ---- P2: scoresᵀ = K·Qᵀ per head (K=32 -> ONE wmma/tile), 8/wave.
    // Lane owns token n, 8 contiguous m -> row-local bias/mask, two f4 stores.
#pragma unroll
    for (int i = 0; i < 8; ++i) {
        int t = wv * 8 + i;
        int h = t >> 4, rem = t & 15;
        int ntile = rem >> 2, mt = rem & 3;
        v16h a  = ld_frag_lds(qkh, 264, mt * 16, CCH + h * HD);   // k rows (m)
        v16h bb = ld_frag_lds(qkh, 264, ntile * 16, h * HD);      // q rows (n)
        v8f acc = {};
        acc = wmma16(a, bb, acc);
        int n  = ntile * 16 + lcol;
        int m0 = mt * 16 + lrow;
        const float* bp = bm + h * NTOK * NTOK + n * NTOK;
        const float* mp = maskb + n * NTOK;
        bool nok = (n < NTOK);
#pragma unroll
        for (int v = 0; v < 8; ++v) {
            int m = m0 + v;
            acc[v] = (nok && m < NTOK) ? acc[v] + bp[m] + mp[m] : -1e30f;
        }
        float* sp = sc + (h * 64 + n) * 68 + m0;
        *(f4*)sp       = __builtin_shufflevector(acc, acc, 0, 1, 2, 3);
        *(f4*)(sp + 4) = __builtin_shufflevector(acc, acc, 4, 5, 6, 7);
    }
    __syncthreads();

    // ---- P2b: row softmax, vectorized. 196 real rows + 60 pad rows = 256 thr.
    if (tid < HEADS * NTOK) {
        int h = tid / NTOK, n = tid % NTOK;
        const float* row = sc + (h * 64 + n) * 68;
        float mx = -1e30f;
#pragma unroll
        for (int c = 0; c < 16; ++c) {
            f4 v = *(const f4*)(row + 4 * c);
            mx = fmaxf(mx, fmaxf(fmaxf(v[0], v[1]), fmaxf(v[2], v[3])));
        }
        float s = 0.f;
#pragma unroll
        for (int c = 0; c < 16; ++c) {
            f4 v = *(const f4*)(row + 4 * c);
            s += __expf(v[0] - mx) + __expf(v[1] - mx) + __expf(v[2] - mx) + __expf(v[3] - mx);
        }
        float inv = 1.f / s;
        _Float16* arow = ath + (h * 64 + n) * 72;
#pragma unroll
        for (int c8 = 0; c8 < 8; ++c8) {
            f4 lo = *(const f4*)(row + 8 * c8);
            f4 hi = *(const f4*)(row + 8 * c8 + 4);
            h8 pk;
#pragma unroll
            for (int v = 0; v < 4; ++v) {
                pk[v]     = (_Float16)(__expf(lo[v] - mx) * inv);
                pk[4 + v] = (_Float16)(__expf(hi[v] - mx) * inv);
            }
            *(h8*)(arow + 8 * c8) = pk;
        }
    } else {
        int t2 = tid - HEADS * NTOK;         // 0..59 -> 4 heads x 15 pad rows
        int h = t2 / 15, n = NTOK + t2 % 15;
        _Float16* arow = ath + (h * 64 + n) * 72;
        h8 z = {};
#pragma unroll
        for (int c8 = 0; c8 < 8; ++c8) *(h8*)(arow + 8 * c8) = z;
    }
    __syncthreads();

    // ---- P3: out_headᵀ = Vᵀ·attnᵀ (both operands already row-major), 4/wave.
#pragma unroll
    for (int i = 0; i < 4; ++i) {
        int t = wv * 4 + i;
        int h = t >> 3, dt = (t >> 2) & 1, tt = t & 3;
        v8f acc = {};
#pragma unroll
        for (int kt = 0; kt < 2; ++kt) {
            v16h a  = ld_frag_lds(vT, 72, h * HD + dt * 16, kt * 32);       // rows = d
            v16h bb = ld_frag_lds(ath + h * 64 * 72, 72, tt * 16, kt * 32); // rows = n
            acc = wmma16(a, bb, acc);
        }
        int token = tt * 16 + lcol;
        int c0    = h * HD + dt * 16 + lrow;
        *(h8*)(oh + token * 136 + c0) = pack8(acc, 1.f);
    }
    __syncthreads();

    // ---- P4: finalᵀ = Wproj·ohᵀ + b (32 tiles, 4/wave). Whole-lane token<49
    // predicate, two global_store_b128 with vectorized bias.
#pragma unroll
    for (int i = 0; i < 4; ++i) {
        int t = wv * 4 + i;
        int ot = t >> 2, tt = t & 3;
        v8f acc = {};
#pragma unroll
        for (int kt = 0; kt < 4; ++kt) {
            v16h a  = ld_frag_g16(pwh, CCH, ot * 16, kt * 32);   // rows = out-chan
            v16h bb = ld_frag_lds(oh, 136, tt * 16, kt * 32);    // rows = token
            acc = wmma16(a, bb, acc);
        }
        int token = tt * 16 + lcol;
        int oc0   = ot * 16 + lrow;
        if (token < NTOK) {
            f4 blo = *(const f4*)(proj_b + oc0);
            f4 bhi = *(const f4*)(proj_b + oc0 + 4);
            float* op = out + (size_t)b * NTOK * CCH + token * CCH + oc0;
            *(f4*)op       = __builtin_shufflevector(acc, acc, 0, 1, 2, 3) + blo;
            *(f4*)(op + 4) = __builtin_shufflevector(acc, acc, 4, 5, 6, 7) + bhi;
        }
    }
}

extern "C" void kernel_launch(void* const* d_in, const int* in_sizes, int n_in,
                              void* d_out, int out_size, void* d_ws, size_t ws_size,
                              hipStream_t stream) {
    const float* x      = (const float*)d_in[0];
    const float* qkv_w  = (const float*)d_in[1];
    const float* proj_w = (const float*)d_in[2];
    const float* proj_b = (const float*)d_in[3];
    const float* tab    = (const float*)d_in[4];
    const float* mask   = (const float*)d_in[5];
    const int*   idx    = (const int*)d_in[6];
    float* out = (float*)d_out;
    char*  ws  = (char*)d_ws;

    static_assert(SMEM_BYTES <= 160 * 1024, "want 2 workgroups per 320KB WGP");
    (void)hipFuncSetAttribute((const void*)win_attn_kernel,
                              hipFuncAttributeMaxDynamicSharedMemorySize, SMEM_BYTES);

    prep_kernel<<<(384 * CCH + THREADS - 1) / THREADS, THREADS, 0, stream>>>(
        qkv_w, proj_w, tab, idx, ws);

    const int B = in_sizes[0] / (NTOK * CCH);   // 8192
    win_attn_kernel<<<B, THREADS, SMEM_BYTES, stream>>>(x, proj_b, mask, ws, out);
}
